// InverseSTFT_9199819948645
// MI455X (gfx1250) — matrix operand order repaired and checked by reading, not compiled, not run on hardware
//
#include <hip/hip_runtime.h>

#define N_FFTC   1024
#define HOPC     256
#define NBATCH   16
#define NFREQ    513
#define NT       2000
#define K_REAL   1026           // 513 bins * {re,im}, symmetry folded into weights
#define K_PAD    1056           // 33 * 32
#define OUT_PER_B 512256        // N_FFT + HOP*(NT-1) - N_FFT/2
#define BM       128
#define BN       128
#define BK       32
#define NKT      (K_PAD / BK)   // 33
#define BKP      40             // padded LDS row stride in halves (bank-conflict-free)

#define BASIS_BYTES ((size_t)N_FFTC * K_PAD * 2)            // 2,162,688 (256B-aligned)
#define BH_BYTES    ((size_t)NBATCH * NT * K_PAD * 2)       // 67,584,000

typedef __attribute__((ext_vector_type(16))) _Float16 v16h;
typedef __attribute__((ext_vector_type(8)))  float    v8f;

union Frag16 { v16h h; int4 i4[2]; };
union Pack2  { _Float16 h[2]; unsigned u; };

// ---------------------------------------------------------------- basis ----
// Folded iSTFT basis, f16, row-major [w][k], k = 2f + c, zero-padded to K_PAD.
__global__ void build_basis_kernel(_Float16* __restrict__ A) {
  const int w = blockIdx.x;
  for (int k = threadIdx.x; k < K_PAD; k += blockDim.x) {
    float val = 0.0f;
    if (k < K_REAL) {
      const int f = k >> 1;
      const float s = (f == 0 || f == (N_FFTC / 2)) ? 1.0f : 2.0f;
      const int ph = (f * w) & (N_FFTC - 1);
      const float ang = (float)ph * (6.28318530717958647692f / (float)N_FFTC);
      float sn, cs;
      __sincosf(ang, &sn, &cs);
      val = ((k & 1) ? -sn : cs) * (s / (float)N_FFTC);
    }
    A[(size_t)w * K_PAD + k] = (_Float16)val;
  }
}

__global__ void zero_out_kernel(float* __restrict__ p, int n) {
  const int i = blockIdx.x * blockDim.x + threadIdx.x;
  if (i < n) p[i] = 0.0f;
}

// -------------------------------------------------- one-shot B transpose ----
// stft (B,F,T,2) f32  ->  Bh[b][t][k] f16 (k = 2f+c, K_PAD stride, zero pad).
// LDS-tiled so global reads (t-contiguous) and writes (k-contiguous) coalesce.
__global__ __launch_bounds__(256) void convert_b_kernel(
    const float* __restrict__ stft, _Float16* __restrict__ Bh) {
  __shared__ _Float16 Ts[BN][BKP];
  const int tid    = threadIdx.x;
  const int t_base = blockIdx.x * BN;
  const int k0     = blockIdx.y * BK;
  const int b      = blockIdx.z;

  const int s_fi = tid >> 4;                 // 0..15 freq in slab
  const int s_tg = tid & 15;                 // 8 t's each
  const int t_g  = t_base + s_tg * 8;
  const int t_c  = (t_g <= NT - 8) ? t_g : (NT - 8);
  const int f_raw = (k0 >> 1) + s_fi;
  const int f = (f_raw < NFREQ) ? f_raw : (NFREQ - 1);
  const float keep = (f_raw < NFREQ && t_g < NT) ? 1.0f : 0.0f;

  const float4* src = (const float4*)(stft + (((size_t)b * NFREQ + f) * NT + t_c) * 2);
  #pragma unroll
  for (int j = 0; j < 4; ++j) {
    const float4 v = src[j];
    Pack2 p0, p1;
    p0.h[0] = (_Float16)(v.x * keep); p0.h[1] = (_Float16)(v.y * keep);
    p1.h[0] = (_Float16)(v.z * keep); p1.h[1] = (_Float16)(v.w * keep);
    *(unsigned*)&Ts[s_tg * 8 + 2 * j][2 * s_fi]     = p0.u;
    *(unsigned*)&Ts[s_tg * 8 + 2 * j + 1][2 * s_fi] = p1.u;
  }
  __syncthreads();

  const int row = tid >> 1;
  const int off = (tid & 1) * 16;
  const int t_out = t_base + row;
  if (t_out < NT) {
    int4* dst = (int4*)(Bh + ((size_t)b * NT + t_out) * K_PAD + k0 + off);
    dst[0] = *(const int4*)&Ts[row][off];
    dst[1] = *(const int4*)&Ts[row][off + 8];
  }
}

// ------------------------------------------------------------- async ops ----
static __device__ __forceinline__ void async_cp_32B(unsigned lds_dst, const void* gsrc) {
  asm volatile(
      "global_load_async_to_lds_b128 %0, %1, off\n\t"
      "global_load_async_to_lds_b128 %0, %1, off offset:16"
      :: "v"(lds_dst), "v"(gsrc) : "memory");
}
static __device__ __forceinline__ void wait_async0() {
  asm volatile("s_wait_asynccnt 0x0" ::: "memory");
}

// ------------------------------------------------ fast GEMM (all-async) ----
__global__ __launch_bounds__(256) void istft_gemm_async_kernel(
    const _Float16* __restrict__ basis,
    const _Float16* __restrict__ Bh,
    float* __restrict__ out) {
  __shared__ _Float16 As[2][BM][BKP];
  __shared__ _Float16 Bs[2][BN][BKP];

  const int tid    = threadIdx.x;
  const int b      = blockIdx.z;
  const int m_base = blockIdx.y * BM;
  const int t_base = blockIdx.x * BN;

  const int lane   = tid & 31;
  const int waveid = tid >> 5;
  const int wave_m = waveid >> 2;               // 0..1 -> 64 rows
  const int wave_n = waveid & 3;                // 0..3 -> 32 cols
  const int lh     = lane & 15;
  const int sel    = lane >> 4;

  // staging: each thread copies 32B of A and 32B of B per K-step
  const int srow  = tid >> 1;                   // 0..127
  const int soff  = (tid & 1) * 16;             // halves
  const int t_src = (t_base + srow < NT) ? (t_base + srow) : (NT - 1); // clamp; extra cols discarded
  const _Float16* a_src0 = basis + (size_t)(m_base + srow) * K_PAD + soff;
  const _Float16* b_src0 = Bh + ((size_t)b * NT + t_src) * K_PAD + soff;

  v8f acc[4][2];
  #pragma unroll
  for (int mi = 0; mi < 4; ++mi)
    #pragma unroll
    for (int ni = 0; ni < 2; ++ni)
      #pragma unroll
      for (int r = 0; r < 8; ++r)
        acc[mi][ni][r] = 0.0f;

  auto stage = [&](int buf, int k0) {
    async_cp_32B((unsigned)(size_t)&As[buf][srow][soff], a_src0 + k0);
    async_cp_32B((unsigned)(size_t)&Bs[buf][srow][soff], b_src0 + k0);
  };

  // prologue
  stage(0, 0);
  wait_async0();
  __syncthreads();

  for (int kt = 0; kt < NKT; ++kt) {
    const int cur = kt & 1;
    const int nxt = cur ^ 1;
    const bool pf = (kt + 1) < NKT;
    if (pf) stage(nxt, (kt + 1) * BK);          // in flight during WMMAs

    Frag16 af[4];
    #pragma unroll
    for (int mi = 0; mi < 4; ++mi) {
      const _Float16* base = &As[cur][wave_m * 64 + mi * 16 + lh][sel * 8];
      af[mi].i4[0] = *(const int4*)(base);
      af[mi].i4[1] = *(const int4*)(base + 16);
    }
    Frag16 bf[2];
    #pragma unroll
    for (int ni = 0; ni < 2; ++ni) {
      const _Float16* base = &Bs[cur][wave_n * 32 + ni * 16 + lh][sel * 8];
      bf[ni].i4[0] = *(const int4*)(base);
      bf[ni].i4[1] = *(const int4*)(base + 16);
    }

    #pragma unroll
    for (int mi = 0; mi < 4; ++mi)
      #pragma unroll
      for (int ni = 0; ni < 2; ++ni)
        acc[mi][ni] = __builtin_amdgcn_wmma_f32_16x16x32_f16(
            false, af[mi].h, false, bf[ni].h, (short)0, acc[mi][ni], false, false);

    if (pf) wait_async0();
    __syncthreads();
  }

  // fused overlap-add epilogue; C layout: VGPR r -> M = r + sel*8 ; N = lh
  float* obase = out + (size_t)b * OUT_PER_B;
  const int wrow0 = m_base + wave_m * 64 + sel * 8;
  const int tcol0 = t_base + wave_n * 32 + lh;
  #pragma unroll
  for (int mi = 0; mi < 4; ++mi) {
    #pragma unroll
    for (int ni = 0; ni < 2; ++ni) {
      const int t = tcol0 + ni * 16;
      if (t < NT) {
        #pragma unroll
        for (int r = 0; r < 8; ++r) {
          const int w = wrow0 + mi * 16 + r;
          const int l = w + HOPC * t - N_FFTC / 2;
          if (l >= 0) atomicAdd(obase + l, acc[mi][ni][r]);
        }
      }
    }
  }
}

// --------------------------- fallback GEMM (in-loop conversion, small ws) ----
struct BReg { float4 v[4]; };

__global__ __launch_bounds__(256) void istft_gemm_conv_kernel(
    const _Float16* __restrict__ basis,
    const float* __restrict__ stft,
    float* __restrict__ out) {
  __shared__ _Float16 As[2][BM][BKP];
  __shared__ _Float16 Bs[2][BN][BKP];

  const int tid    = threadIdx.x;
  const int b      = blockIdx.z;
  const int m_base = blockIdx.y * BM;
  const int t_base = blockIdx.x * BN;

  const int lane   = tid & 31;
  const int waveid = tid >> 5;
  const int wave_m = waveid >> 2;
  const int wave_n = waveid & 3;
  const int lh     = lane & 15;
  const int sel    = lane >> 4;

  const int a_row = tid >> 1;
  const int a_off = (tid & 1) * 16;
  const int s_fi  = tid >> 4;
  const int s_tg  = tid & 15;
  const int t_g   = t_base + s_tg * 8;
  const int t_c   = (t_g <= NT - 8) ? t_g : (NT - 8);
  const bool t_ok = (t_g < NT);

  v8f acc[4][2];
  #pragma unroll
  for (int mi = 0; mi < 4; ++mi)
    #pragma unroll
    for (int ni = 0; ni < 2; ++ni)
      #pragma unroll
      for (int r = 0; r < 8; ++r)
        acc[mi][ni][r] = 0.0f;

  auto issue_async_a = [&](int buf, int k0) {
    const _Float16* src = basis + (size_t)(m_base + a_row) * K_PAD + k0 + a_off;
    async_cp_32B((unsigned)(size_t)&As[buf][a_row][a_off], src);
  };
  auto load_b_regs = [&](int k0) -> BReg {
    const int f_raw = (k0 >> 1) + s_fi;
    const int f = (f_raw < NFREQ) ? f_raw : (NFREQ - 1);
    const float4* src = (const float4*)(stft + (((size_t)b * NFREQ + f) * NT + t_c) * 2);
    BReg r;
    #pragma unroll
    for (int j = 0; j < 4; ++j) r.v[j] = src[j];
    return r;
  };
  auto store_b_regs = [&](int buf, const BReg& r, int k0) {
    const int f_raw = (k0 >> 1) + s_fi;
    const float keep = (f_raw < NFREQ && t_ok) ? 1.0f : 0.0f;
    #pragma unroll
    for (int j = 0; j < 4; ++j) {
      const float4 v = r.v[j];
      Pack2 p0, p1;
      p0.h[0] = (_Float16)(v.x * keep); p0.h[1] = (_Float16)(v.y * keep);
      p1.h[0] = (_Float16)(v.z * keep); p1.h[1] = (_Float16)(v.w * keep);
      *(unsigned*)&Bs[buf][s_tg * 8 + 2 * j][2 * s_fi]     = p0.u;
      *(unsigned*)&Bs[buf][s_tg * 8 + 2 * j + 1][2 * s_fi] = p1.u;
    }
  };

  issue_async_a(0, 0);
  {
    BReg r0 = load_b_regs(0);
    store_b_regs(0, r0, 0);
  }
  wait_async0();
  __syncthreads();

  for (int kt = 0; kt < NKT; ++kt) {
    const int cur = kt & 1;
    const int nxt = cur ^ 1;
    const int k0n = (kt + 1) * BK;
    const bool pf = (kt + 1) < NKT;

    BReg breg;
    if (pf) { issue_async_a(nxt, k0n); breg = load_b_regs(k0n); }

    Frag16 af[4];
    #pragma unroll
    for (int mi = 0; mi < 4; ++mi) {
      const _Float16* base = &As[cur][wave_m * 64 + mi * 16 + lh][sel * 8];
      af[mi].i4[0] = *(const int4*)(base);
      af[mi].i4[1] = *(const int4*)(base + 16);
    }
    Frag16 bf[2];
    #pragma unroll
    for (int ni = 0; ni < 2; ++ni) {
      const _Float16* base = &Bs[cur][wave_n * 32 + ni * 16 + lh][sel * 8];
      bf[ni].i4[0] = *(const int4*)(base);
      bf[ni].i4[1] = *(const int4*)(base + 16);
    }
    #pragma unroll
    for (int mi = 0; mi < 4; ++mi)
      #pragma unroll
      for (int ni = 0; ni < 2; ++ni)
        acc[mi][ni] = __builtin_amdgcn_wmma_f32_16x16x32_f16(
            false, af[mi].h, false, bf[ni].h, (short)0, acc[mi][ni], false, false);

    if (pf) { store_b_regs(nxt, breg, k0n); wait_async0(); }
    __syncthreads();
  }

  float* obase = out + (size_t)b * OUT_PER_B;
  const int wrow0 = m_base + wave_m * 64 + sel * 8;
  const int tcol0 = t_base + wave_n * 32 + lh;
  #pragma unroll
  for (int mi = 0; mi < 4; ++mi) {
    #pragma unroll
    for (int ni = 0; ni < 2; ++ni) {
      const int t = tcol0 + ni * 16;
      if (t < NT) {
        #pragma unroll
        for (int r = 0; r < 8; ++r) {
          const int w = wrow0 + mi * 16 + r;
          const int l = w + HOPC * t - N_FFTC / 2;
          if (l >= 0) atomicAdd(obase + l, acc[mi][ni][r]);
        }
      }
    }
  }
}

// Divide by analytic window-sum = #frames t with 256t <= p < 256t+1024
__global__ void wss_scale_kernel(float* __restrict__ out, int total) {
  const int g = blockIdx.x * blockDim.x + threadIdx.x;
  if (g >= total) return;
  const int l = g % OUT_PER_B;
  const int p = l + N_FFTC / 2;
  int tmax = p >> 8;                      if (tmax > NT - 1) tmax = NT - 1;
  int tmin = (p - (N_FFTC - HOPC)) >> 8;  if (tmin < 0) tmin = 0;
  out[g] /= (float)(tmax - tmin + 1);
}

extern "C" void kernel_launch(void* const* d_in, const int* in_sizes, int n_in,
                              void* d_out, int out_size, void* d_ws, size_t ws_size,
                              hipStream_t stream) {
  const float* stft = (const float*)d_in[0];
  float* out = (float*)d_out;
  _Float16* basis = (_Float16*)d_ws;

  build_basis_kernel<<<N_FFTC, 256, 0, stream>>>(basis);

  const int total = NBATCH * OUT_PER_B;
  zero_out_kernel<<<(total + 255) / 256, 256, 0, stream>>>(out, total);

  dim3 grid((NT + BN - 1) / BN, N_FFTC / BM, NBATCH);   // 16 x 8 x 16

  if (ws_size >= BASIS_BYTES + BH_BYTES) {
    // fast path: one-shot f32->f16 k-major transpose, then all-async GEMM
    _Float16* Bh = (_Float16*)((char*)d_ws + BASIS_BYTES);
    dim3 cgrid((NT + BN - 1) / BN, NKT, NBATCH);        // 16 x 33 x 16
    convert_b_kernel<<<cgrid, 256, 0, stream>>>(stft, Bh);
    istft_gemm_async_kernel<<<grid, 256, 0, stream>>>(basis, Bh, out);
  } else {
    // fallback: convert inside the GEMM (only basis lives in d_ws)
    istft_gemm_conv_kernel<<<grid, 256, 0, stream>>>(basis, stft, out);
  }

  wss_scale_kernel<<<(total + 255) / 256, 256, 0, stream>>>(out, total);
}
